// SparseTransformerLayer_32822140076720
// MI455X (gfx1250) — matrix operand order repaired and checked by reading, not compiled, not run on hardware
//
#include <hip/hip_runtime.h>

// ---------------------------------------------------------------------------
// Types for CDNA5 WMMA (wave32, v_wmma_f32_16x16x32_bf16)
// ---------------------------------------------------------------------------
typedef __bf16 bf16;
typedef __attribute__((ext_vector_type(8)))  bf16  bf16x8;
typedef __attribute__((ext_vector_type(16))) bf16  v16bf;
typedef __attribute__((ext_vector_type(8)))  float v8f;

#define DEVI __device__ __forceinline__

union BFPack { v16bf v; bf16x8 h[2]; };
DEVI v16bf make_frag(bf16x8 lo, bf16x8 hi) { BFPack u; u.h[0] = lo; u.h[1] = hi; return u.v; }
DEVI bf16x8 load8(const bf16* p) { return *(const bf16x8*)p; }

// gfx1250 async global->LDS DMA (ASYNCcnt-tracked, no VGPR bounce).
// lds operand = LDS byte address (low 32 bits of flat shared pointer).
DEVI void async_load_b128_to_lds(const bf16* lds_ptr, const bf16* gptr) {
  unsigned lds = (unsigned)(unsigned long long)lds_ptr;
  asm volatile("global_load_async_to_lds_b128 %0, %1, off"
               :: "v"(lds), "v"(gptr) : "memory");
}
DEVI void wait_asynccnt0() { asm volatile("s_wait_asynccnt 0x0" ::: "memory"); }

// ---------------------------------------------------------------------------
// Problem constants (match reference)
// ---------------------------------------------------------------------------
constexpr int Bb  = 4;
constexpr int Ss  = 2048;
constexpr int Dd  = 1024;
constexpr int Hh  = 16;
constexpr int Ee  = 8;
constexpr int Kt  = 2;        // top-k
constexpr int Ff  = 4096;
constexpr int FSs = 2048;
constexpr int Tt  = Bb * Ss;              // 8192 tokens
constexpr int CAPc = (5 * Kt * Tt) / (4 * Ee);  // 1.25 * K * T / E = 2560

// ---------------------------------------------------------------------------
// Generic bf16 WMMA GEMM:  C[M,N] = A[M,K] * B[K,N] (+bias) (+gelu) (+res)
// 128x128 tile / block, 256 threads = 8 waves, each wave 64x32 (4x2 wmma tiles)
// A tile staged via global_load_async_to_lds_b128; B tile transposed in LDS.
// Batched via blockIdx.z with element strides sA/sB/sBias/sC.
// ---------------------------------------------------------------------------
template<bool BIAS, bool GELU, bool RES, bool OUTF, bool OUTB>
__global__ __launch_bounds__(256) void gemm_bf16_kernel(
    const bf16* __restrict__ A,  long sA,
    const bf16* __restrict__ Bw, long sB,
    const float* __restrict__ bias, long sBias,
    const float* __restrict__ res,
    float* __restrict__ Cf, bf16* __restrict__ Cb, long sC,
    int N, int Kd)
{
  constexpr int LDA = 40;                       // bf16 elements; 80B rows keep 16B align
  __shared__ bf16 As[128 * LDA];                // A tile, row-major [m][k]
  __shared__ bf16 Bs[128 * LDA];                // B tile, TRANSPOSED [n][k]

  const int z = blockIdx.z;
  A  += (long)z * sA;
  Bw += (long)z * sB;
  if (BIAS) bias += (long)z * sBias;
  const long coff = (long)z * sC;

  const int tid  = threadIdx.x;
  const int lane = tid & 31;
  const int wave = tid >> 5;
  const int wm   = (wave >> 2) * 64;            // wave M origin in tile
  const int wn   = (wave & 3) * 32;             // wave N origin in tile
  const int m0   = blockIdx.y * 128;
  const int n0   = blockIdx.x * 128;
  const int ml   = lane & 15;
  const int kh   = lane >> 4;

  v8f acc[4][2];
#pragma unroll
  for (int i = 0; i < 4; ++i)
#pragma unroll
    for (int j = 0; j < 2; ++j)
#pragma unroll
      for (int r = 0; r < 8; ++r) acc[i][j][r] = 0.f;

  for (int k0 = 0; k0 < Kd; k0 += 32) {
    // ---- stage A (128x32) via async DMA straight into LDS ----
#pragma unroll
    for (int i = 0; i < 2; ++i) {
      int c  = tid + i * 256;                   // 0..511 chunks of 8 bf16 (16B)
      int r  = c >> 2;
      int kc = (c & 3) * 8;
      async_load_b128_to_lds(As + r * LDA + kc,
                             A + (long)(m0 + r) * Kd + (k0 + kc));
    }
    // ---- stage B (32x128) transposed -> Bs[n][k] ----
#pragma unroll
    for (int i = 0; i < 2; ++i) {
      int c  = tid + i * 256;
      int kk = c >> 4;
      int nc = (c & 15) * 8;
      bf16x8 v = load8(Bw + (long)(k0 + kk) * N + (n0 + nc));
#pragma unroll
      for (int j = 0; j < 8; ++j) Bs[(nc + j) * LDA + kk] = v[j];
    }
    if (k0 + 32 < Kd)  // gfx1250 global_prefetch_b8 for next B tile
      __builtin_prefetch(Bw + (long)(k0 + 32 + (tid >> 4)) * N + n0, 0, 0);
    wait_asynccnt0();      // drain this wave's async LDS writes before barrier
    __syncthreads();

    // ---- fragments ----
    // A layout (16-bit 16x32): lane = M + 16*khalf; elems = K[kh*8..+7], K[16+kh*8..+7]
    v16bf af[4];
#pragma unroll
    for (int i = 0; i < 4; ++i) {
      const bf16* pa = As + (wm + i * 16 + ml) * LDA + kh * 8;
      af[i] = make_frag(load8(pa), load8(pa + 16));
    }
    // B layout (16-bit 32x16): lane = N + 16*khalf; elems = K[kh*8..+7], K[16+kh*8..+7]
    v16bf bfb[2];
#pragma unroll
    for (int j = 0; j < 2; ++j) {
      const bf16* pb = Bs + (wn + j * 16 + ml) * LDA + kh * 8;
      bfb[j] = make_frag(load8(pb), load8(pb + 16));
    }
#pragma unroll
    for (int i = 0; i < 4; ++i)
#pragma unroll
      for (int j = 0; j < 2; ++j)
        acc[i][j] = __builtin_amdgcn_wmma_f32_16x16x32_bf16(
            false, af[i], false, bfb[j], (short)0, acc[i][j], false, false);
    __syncthreads();
  }

  // ---- epilogue (C layout: VGPR r -> row r + 8*khalf, col = ml) ----
#pragma unroll
  for (int i = 0; i < 4; ++i)
#pragma unroll
    for (int j = 0; j < 2; ++j) {
      int col  = n0 + wn + j * 16 + ml;
      float bv = BIAS ? bias[col] : 0.f;
#pragma unroll
      for (int r = 0; r < 8; ++r) {
        int row  = m0 + wm + i * 16 + kh * 8 + r;
        long idx = (long)row * N + col;
        float v  = acc[i][j][r] + bv;
        if (GELU) v = 0.5f * v * (1.f + erff(v * 0.70710678118654752f));
        if (RES)  v += res[idx];
        if (OUTF) Cf[coff + idx] = v;
        if (OUTB) Cb[coff + idx] = (bf16)v;
      }
    }
}

// ---------------------------------------------------------------------------
// Flash attention, WMMA QK^T and PV. Q/K/V layout: [B*S, H*64] bf16.
// Block = 256 thr (8 waves). Each wave: 16 q-rows; loop keys in blocks of 32.
// ---------------------------------------------------------------------------
__global__ __launch_bounds__(256) void attn_kernel(
    const bf16* __restrict__ Q, const bf16* __restrict__ Kb,
    const bf16* __restrict__ V, const unsigned char* __restrict__ amask,
    bf16* __restrict__ O)
{
  constexpr int LDV = 40;
  __shared__ bf16 Vt[64 * LDV];        // V tile transposed: [feat][key]
  __shared__ bf16 Pw[8 * 16 * LDV];    // per-wave P scratch (C-layout -> A-layout)

  const int tid = threadIdx.x, lane = tid & 31, wave = tid >> 5;
  const int ml = lane & 15, kh = lane >> 4;
  const int b  = blockIdx.z, h = blockIdx.y;
  const int q0 = blockIdx.x * 128 + wave * 16;
  const long rowbase = ((long)b * Ss) * Dd + h * 64;

  // Q fragments (16 x 64) = two 16x32 A-fragments, loaded once
  v16bf aq[2];
#pragma unroll
  for (int kk = 0; kk < 2; ++kk) {
    const bf16* pq = Q + rowbase + (long)(q0 + ml) * Dd + kk * 32 + kh * 8;
    aq[kk] = make_frag(load8(pq), load8(pq + 16));
  }

  float rm[8], rl[8];
  v8f ao[4];
#pragma unroll
  for (int r = 0; r < 8; ++r) { rm[r] = -3.0e38f; rl[r] = 0.f; }
#pragma unroll
  for (int nt = 0; nt < 4; ++nt)
#pragma unroll
    for (int r = 0; r < 8; ++r) ao[nt][r] = 0.f;

  bf16* Pme = Pw + wave * 16 * LDV;

  for (int kb = 0; kb < Ss; kb += 32) {
    __syncthreads();
    { // cooperative stage of V tile (32 keys x 64 feats), transposed
      int r  = tid >> 3;
      int fc = (tid & 7) * 8;
      bf16x8 v = load8(V + rowbase + (long)(kb + r) * Dd + fc);
#pragma unroll
      for (int j = 0; j < 8; ++j) Vt[(fc + j) * LDV + r] = v[j];
    }
    __syncthreads();

    // ---- scores S = Q K^T (16 x 32) ----
    v8f sc[2];
#pragma unroll
    for (int nt = 0; nt < 2; ++nt) {
#pragma unroll
      for (int r = 0; r < 8; ++r) sc[nt][r] = 0.f;
#pragma unroll
      for (int kk = 0; kk < 2; ++kk) {
        // B-frag of K^T straight from global: key = column, features contiguous
        const bf16* pk = Kb + rowbase + (long)(kb + nt * 16 + ml) * Dd + kk * 32 + kh * 8;
        v16bf bk = make_frag(load8(pk), load8(pk + 16));
        sc[nt] = __builtin_amdgcn_wmma_f32_16x16x32_bf16(
            false, aq[kk], false, bk, (short)0, sc[nt], false, false);
      }
    }
    // scale + mask
#pragma unroll
    for (int nt = 0; nt < 2; ++nt) {
      bool msk = amask[(long)b * Ss + kb + nt * 16 + ml] != 0;
#pragma unroll
      for (int r = 0; r < 8; ++r) {
        float s = sc[nt][r] * 0.125f;   // 1/sqrt(64)
        sc[nt][r] = msk ? -3.0e38f : s;
      }
    }
    // ---- online softmax (row = r + 8*kh, cols spread over 16 lanes) ----
    float mnew[8], scl[8], ps[8];
#pragma unroll
    for (int r = 0; r < 8; ++r) mnew[r] = fmaxf(sc[0][r], sc[1][r]);
#pragma unroll
    for (int off = 1; off < 16; off <<= 1)
#pragma unroll
      for (int r = 0; r < 8; ++r) mnew[r] = fmaxf(mnew[r], __shfl_xor(mnew[r], off, 32));
#pragma unroll
    for (int r = 0; r < 8; ++r) {
      mnew[r] = fmaxf(mnew[r], rm[r]);
      scl[r]  = __expf(rm[r] - mnew[r]);
      ps[r]   = 0.f;
    }
#pragma unroll
    for (int nt = 0; nt < 2; ++nt)
#pragma unroll
      for (int r = 0; r < 8; ++r) {
        float pv = __expf(sc[nt][r] - mnew[r]);
        ps[r] += pv;
        Pme[(r + kh * 8) * LDV + nt * 16 + ml] = (bf16)pv;
      }
#pragma unroll
    for (int off = 1; off < 16; off <<= 1)
#pragma unroll
      for (int r = 0; r < 8; ++r) ps[r] += __shfl_xor(ps[r], off, 32);
#pragma unroll
    for (int r = 0; r < 8; ++r) {
      rl[r] = rl[r] * scl[r] + ps[r];
      rm[r] = mnew[r];
    }
#pragma unroll
    for (int nt = 0; nt < 4; ++nt)
#pragma unroll
      for (int r = 0; r < 8; ++r) ao[nt][r] *= scl[r];

    // ---- P (A-layout from LDS) x V ----
    const bf16* pp = Pme + ml * LDV + kh * 8;
    v16bf pf = make_frag(load8(pp), load8(pp + 16));
#pragma unroll
    for (int nt = 0; nt < 4; ++nt) {
      const bf16* pv = Vt + (nt * 16 + ml) * LDV + kh * 8;
      v16bf bv = make_frag(load8(pv), load8(pv + 16));
      ao[nt] = __builtin_amdgcn_wmma_f32_16x16x32_bf16(
          false, pf, false, bv, (short)0, ao[nt], false, false);
    }
  }

  // ---- epilogue ----
#pragma unroll
  for (int r = 0; r < 8; ++r) rl[r] = 1.f / fmaxf(rl[r], 1e-20f);
#pragma unroll
  for (int nt = 0; nt < 4; ++nt)
#pragma unroll
    for (int r = 0; r < 8; ++r) {
      int m = r + kh * 8;
      O[rowbase + (long)(q0 + m) * Dd + nt * 16 + ml] = (bf16)(ao[nt][r] * rl[r]);
    }
}

// ---------------------------------------------------------------------------
// LayerNorm (one block per row of D=1024). Writes fp32 (optional) + bf16.
// ---------------------------------------------------------------------------
__global__ __launch_bounds__(256) void ln_kernel(
    const float* __restrict__ x, const float* __restrict__ g, const float* __restrict__ bt,
    float* __restrict__ outf, bf16* __restrict__ outb)
{
  __shared__ float s1[256], s2[256];
  const long row = blockIdx.x;
  const int tid = threadIdx.x;
  const float* xr = x + row * Dd;
  float v[4], a = 0.f, q = 0.f;
#pragma unroll
  for (int i = 0; i < 4; ++i) { v[i] = xr[tid + i * 256]; a += v[i]; q += v[i] * v[i]; }
  s1[tid] = a; s2[tid] = q; __syncthreads();
  for (int w = 128; w > 0; w >>= 1) {
    if (tid < w) { s1[tid] += s1[tid + w]; s2[tid] += s2[tid + w]; }
    __syncthreads();
  }
  float mean = s1[0] * (1.f / Dd);
  float var  = s2[0] * (1.f / Dd) - mean * mean;
  float rstd = rsqrtf(var + 1e-5f);
#pragma unroll
  for (int i = 0; i < 4; ++i) {
    int d = tid + i * 256;
    float y = (v[i] - mean) * rstd * g[d] + bt[d];
    if (outf) outf[row * Dd + d] = y;
    outb[row * Dd + d] = (bf16)y;
  }
}

// ---------------------------------------------------------------------------
// Router: logits (fp32), softmax over E=8, top-2, renormalized gates.
// ---------------------------------------------------------------------------
__global__ __launch_bounds__(256) void router_kernel(
    const float* __restrict__ xn, const float* __restrict__ wr,
    float* __restrict__ topv, int* __restrict__ topi)
{
  __shared__ float part[256];
  __shared__ float lg[8];
  const int t = blockIdx.x, tid = threadIdx.x;
  const int e = tid & 7, g = tid >> 3;
  float a = 0.f;
  for (int d = g; d < Dd; d += 32) a += xn[(long)t * Dd + d] * wr[d * Ee + e];
  part[tid] = a; __syncthreads();
  if (tid < 8) {
    float s = 0.f;
    for (int gg = 0; gg < 32; ++gg) s += part[tid + 8 * gg];
    lg[tid] = s;
  }
  __syncthreads();
  if (tid == 0) {
    float mx = lg[0];
    for (int i = 1; i < 8; ++i) mx = fmaxf(mx, lg[i]);
    float pr[8], sum = 0.f;
    for (int i = 0; i < 8; ++i) { pr[i] = __expf(lg[i] - mx); sum += pr[i]; }
    float inv = 1.f / sum;
    for (int i = 0; i < 8; ++i) pr[i] *= inv;
    int i0 = 0; for (int i = 1; i < 8; ++i) if (pr[i] > pr[i0]) i0 = i;
    int i1 = -1;
    for (int i = 0; i < 8; ++i) { if (i == i0) continue; if (i1 < 0 || pr[i] > pr[i1]) i1 = i; }
    float v0 = pr[i0], v1 = pr[i1], rs = 1.f / (v0 + v1);
    topv[t * 2 + 0] = v0 * rs; topv[t * 2 + 1] = v1 * rs;
    topi[t * 2 + 0] = i0;      topi[t * 2 + 1] = i1;
  }
}

__global__ void fill_i32(int* __restrict__ p, int v, int n) {
  int i = blockIdx.x * blockDim.x + threadIdx.x;
  if (i < n) p[i] = v;
}

// ---------------------------------------------------------------------------
// Capacity scan: slot-major cumsum per expert (one wave per expert, ballot).
// kt = slot*T + t, matches reference flat = oh.transpose(1,0,2).
// ---------------------------------------------------------------------------
__global__ void moe_scan_kernel(
    const int* __restrict__ topi, const float* __restrict__ topv,
    int* __restrict__ slot_token, int* __restrict__ pos_arr, float* __restrict__ wgt)
{
  const int e = blockIdx.x;
  const int lane = threadIdx.x;
  int cnt = 0;
  for (int base = 0; base < Kt * Tt; base += 32) {
    int kt = base + lane;
    int slot = kt / Tt;
    int t = kt % Tt;
    int ei = topi[t * 2 + slot];
    bool match = (ei == e);
    unsigned mask = (unsigned)__ballot(match);
    int pos = cnt + __popc(mask & ((1u << lane) - 1u));
    if (match) {
      bool keep = pos < CAPc;
      pos_arr[kt] = keep ? pos : (CAPc - 1);
      wgt[kt] = keep ? topv[t * 2 + slot] : 0.f;
      if (keep) slot_token[e * CAPc + pos] = t;
    }
    cnt += __popc(mask);
  }
}

// Dispatch: gather kept tokens into [E*CAP, D] (zeros for empty slots).
__global__ __launch_bounds__(256) void dispatch_kernel(
    const int* __restrict__ slot_token, const bf16* __restrict__ xnb,
    bf16* __restrict__ disp)
{
  const long row = blockIdx.x;
  const int tid = threadIdx.x;
  const int tok = slot_token[row];
#pragma unroll
  for (int i = 0; i < 4; ++i) {
    int d = tid + i * 256;
    disp[row * Dd + d] = (tok >= 0) ? xnb[(long)tok * Dd + d] : (bf16)0.f;
  }
}

// Combine: out = x1 + shared_y + sum_k gate_k * eo[expert,pos]
__global__ __launch_bounds__(256) void combine_kernel(
    const float* __restrict__ x1, const float* __restrict__ shared_y,
    const float* __restrict__ eo, const int* __restrict__ topi,
    const int* __restrict__ pos_arr, const float* __restrict__ wgt,
    float* __restrict__ out)
{
  const long t = blockIdx.x;
  const int tid = threadIdx.x;
  int e0 = topi[t * 2 + 0], e1 = topi[t * 2 + 1];
  int p0 = pos_arr[t],      p1 = pos_arr[Tt + t];
  float w0 = wgt[t],        w1 = wgt[Tt + t];
  const float* r0 = eo + ((long)e0 * CAPc + p0) * Dd;
  const float* r1 = eo + ((long)e1 * CAPc + p1) * Dd;
#pragma unroll
  for (int i = 0; i < 4; ++i) {
    int d = tid + i * 256;
    long idx = t * Dd + d;
    out[idx] = x1[idx] + shared_y[idx] + w0 * r0[d] + w1 * r1[d];
  }
}

__global__ void cast_f32_bf16(const float* __restrict__ in, bf16* __restrict__ out, long n) {
  long i = (long)blockIdx.x * blockDim.x + threadIdx.x;
  long stride = (long)gridDim.x * blockDim.x;
  for (; i < n; i += stride) out[i] = (bf16)in[i];
}

// ---------------------------------------------------------------------------
// Orchestration
// ---------------------------------------------------------------------------
extern "C" void kernel_launch(void* const* d_in, const int* in_sizes, int n_in,
                              void* d_out, int out_size, void* d_ws, size_t ws_size,
                              hipStream_t stream) {
  (void)in_sizes; (void)n_in; (void)out_size; (void)ws_size;

  const float* x     = (const float*)d_in[0];
  const unsigned char* amask = (const unsigned char*)d_in[1];
  const float* ln1_g = (const float*)d_in[2];
  const float* ln1_b = (const float*)d_in[3];
  const float* wq = (const float*)d_in[4];  const float* bq = (const float*)d_in[5];
  const float* wk = (const float*)d_in[6];  const float* bk = (const float*)d_in[7];
  const float* wv = (const float*)d_in[8];  const float* bv = (const float*)d_in[9];
  const float* wo = (const float*)d_in[10]; const float* bo = (const float*)d_in[11];
  const float* ln2_g = (const float*)d_in[12]; const float* ln2_b = (const float*)d_in[13];
  const float* w_router = (const float*)d_in[14];
  const float* w1 = (const float*)d_in[15]; const float* b1 = (const float*)d_in[16];
  const float* w2 = (const float*)d_in[17]; const float* b2 = (const float*)d_in[18];
  const float* ws1 = (const float*)d_in[19]; const float* bs1 = (const float*)d_in[20];
  const float* ws2 = (const float*)d_in[21]; const float* bs2 = (const float*)d_in[22];
  float* out = (float*)d_out;

  // bump allocator over d_ws
  char* p = (char*)d_ws;
  auto alloc = [&](size_t bytes) -> char* {
    char* r = p; p += (bytes + 255) & ~(size_t)255; return r;
  };
  auto allocB = [&](long n) -> bf16*  { return (bf16*) alloc((size_t)n * 2); };
  auto allocF = [&](long n) -> float* { return (float*)alloc((size_t)n * 4); };
  auto allocI = [&](long n) -> int*   { return (int*)  alloc((size_t)n * 4); };

  // bf16 weight copies (144 MB total -> fits global L2)
  bf16* wqb  = allocB((long)Dd * Dd);
  bf16* wkb  = allocB((long)Dd * Dd);
  bf16* wvb  = allocB((long)Dd * Dd);
  bf16* wob  = allocB((long)Dd * Dd);
  bf16* w1b  = allocB((long)Ee * Dd * Ff);
  bf16* w2b  = allocB((long)Ee * Ff * Dd);
  bf16* ws1b = allocB((long)Dd * FSs);
  bf16* ws2b = allocB((long)FSs * Dd);
  // activations
  bf16* xn1b  = allocB((long)Tt * Dd);
  bf16* qb    = allocB((long)Tt * Dd);
  bf16* kbuf  = allocB((long)Tt * Dd);
  bf16* vbuf  = allocB((long)Tt * Dd);
  bf16* attnb = allocB((long)Tt * Dd);
  float* x1   = allocF((long)Tt * Dd);
  float* xn2f = allocF((long)Tt * Dd);
  bf16* xn2b  = allocB((long)Tt * Dd);
  float* topv = allocF((long)Tt * 2);
  int*   topi = allocI((long)Tt * 2);
  int*   pos_arr = allocI((long)Kt * Tt);
  float* wgt     = allocF((long)Kt * Tt);
  int*   slot_token = allocI((long)Ee * CAPc);
  bf16*  disp = allocB((long)Ee * CAPc * Dd);
  bf16*  hb   = allocB((long)Ee * CAPc * Ff);
  float* eo   = allocF((long)Ee * CAPc * Dd);
  bf16*  hsb  = allocB((long)Tt * FSs);
  float* sharedy = allocF((long)Tt * Dd);

  const dim3 blk(256);

  // ---- weight casts ----
  cast_f32_bf16<<<2048, blk, 0, stream>>>(wq,  wqb,  (long)Dd * Dd);
  cast_f32_bf16<<<2048, blk, 0, stream>>>(wk,  wkb,  (long)Dd * Dd);
  cast_f32_bf16<<<2048, blk, 0, stream>>>(wv,  wvb,  (long)Dd * Dd);
  cast_f32_bf16<<<2048, blk, 0, stream>>>(wo,  wob,  (long)Dd * Dd);
  cast_f32_bf16<<<4096, blk, 0, stream>>>(w1,  w1b,  (long)Ee * Dd * Ff);
  cast_f32_bf16<<<4096, blk, 0, stream>>>(w2,  w2b,  (long)Ee * Ff * Dd);
  cast_f32_bf16<<<2048, blk, 0, stream>>>(ws1, ws1b, (long)Dd * FSs);
  cast_f32_bf16<<<2048, blk, 0, stream>>>(ws2, ws2b, (long)FSs * Dd);

  // ---- LN1 -> QKV GEMMs ----
  ln_kernel<<<Tt, blk, 0, stream>>>(x, ln1_g, ln1_b, nullptr, xn1b);
  gemm_bf16_kernel<true, false, false, false, true><<<dim3(Dd / 128, Tt / 128, 1), blk, 0, stream>>>(
      xn1b, 0, wqb, 0, bq, 0, nullptr, nullptr, qb, 0, Dd, Dd);
  gemm_bf16_kernel<true, false, false, false, true><<<dim3(Dd / 128, Tt / 128, 1), blk, 0, stream>>>(
      xn1b, 0, wkb, 0, bk, 0, nullptr, nullptr, kbuf, 0, Dd, Dd);
  gemm_bf16_kernel<true, false, false, false, true><<<dim3(Dd / 128, Tt / 128, 1), blk, 0, stream>>>(
      xn1b, 0, wvb, 0, bv, 0, nullptr, nullptr, vbuf, 0, Dd, Dd);

  // ---- attention ----
  attn_kernel<<<dim3(Ss / 128, Hh, Bb), blk, 0, stream>>>(qb, kbuf, vbuf, amask, attnb);

  // ---- WO + residual -> x1 ----
  gemm_bf16_kernel<true, false, true, true, false><<<dim3(Dd / 128, Tt / 128, 1), blk, 0, stream>>>(
      attnb, 0, wob, 0, bo, 0, x, x1, nullptr, 0, Dd, Dd);

  // ---- LN2, router, capacity scan, dispatch ----
  ln_kernel<<<Tt, blk, 0, stream>>>(x1, ln2_g, ln2_b, xn2f, xn2b);
  router_kernel<<<Tt, blk, 0, stream>>>(xn2f, w_router, topv, topi);
  fill_i32<<<(Ee * CAPc + 255) / 256, blk, 0, stream>>>(slot_token, -1, Ee * CAPc);
  moe_scan_kernel<<<Ee, 32, 0, stream>>>(topi, topv, slot_token, pos_arr, wgt);
  dispatch_kernel<<<Ee * CAPc, blk, 0, stream>>>(slot_token, xn2b, disp);

  // ---- expert FFN (batched over E) ----
  gemm_bf16_kernel<true, true, false, false, true><<<dim3(Ff / 128, CAPc / 128, Ee), blk, 0, stream>>>(
      disp, (long)CAPc * Dd, w1b, (long)Dd * Ff, b1, (long)Ff,
      nullptr, nullptr, hb, (long)CAPc * Ff, Ff, Dd);
  gemm_bf16_kernel<true, false, false, true, false><<<dim3(Dd / 128, CAPc / 128, Ee), blk, 0, stream>>>(
      hb, (long)CAPc * Ff, w2b, (long)Ff * Dd, b2, (long)Dd,
      nullptr, eo, nullptr, (long)CAPc * Dd, Dd, Ff);

  // ---- shared expert ----
  gemm_bf16_kernel<true, true, false, false, true><<<dim3(FSs / 128, Tt / 128, 1), blk, 0, stream>>>(
      xn2b, 0, ws1b, 0, bs1, 0, nullptr, nullptr, hsb, 0, FSs, Dd);
  gemm_bf16_kernel<true, false, false, true, false><<<dim3(Dd / 128, Tt / 128, 1), blk, 0, stream>>>(
      hsb, 0, ws2b, 0, bs2, 0, nullptr, sharedy, nullptr, 0, Dd, FSs);

  // ---- combine + residual ----
  combine_kernel<<<Tt, blk, 0, stream>>>(x1, sharedy, eo, topi, pos_arr, wgt, out);
}
